// PoxelGCN_55886114456062
// MI455X (gfx1250) — matrix operand order repaired and checked by reading, not compiled, not run on hardware
//
#include <hip/hip_runtime.h>
#include <cstdint>

#define KNN_K 6
#define HB 256
#define BB 32
#define NEGS 0.2f
#define BNEPS 1e-5f
#define NEGINF_KEY 0x007fffffu

typedef _Float16 half_t;
typedef _Float16 v16h __attribute__((ext_vector_type(16)));
typedef float v8f __attribute__((ext_vector_type(8)));

// order-preserving float -> uint key (for atomicMax on floats / bitonic sort)
__device__ __forceinline__ unsigned fkey(float f) {
  unsigned u = __float_as_uint(f);
  return (u & 0x80000000u) ? ~u : (u | 0x80000000u);
}
__device__ __forceinline__ float funkey(unsigned k) {
  unsigned u = (k & 0x80000000u) ? (k & 0x7fffffffu) : ~k;
  return __uint_as_float(u);
}

__device__ __forceinline__ void decodeEdge(const int* __restrict__ nbr, int b, int e,
                                           int N, int& src, int& dst) {
  int NK = N * KNN_K;
  int idx = (e < NK) ? e : e - NK;
  int i = idx / KNN_K;
  int m = nbr[(size_t)b * NK + idx];
  if (e < NK) { src = m; dst = i; } else { src = i; dst = m; }
}

// ---------------- fills ----------------
__global__ void fill_f32(float* p, float v, long long n) {
  long long t = (long long)blockIdx.x * 256 + threadIdx.x;
  if (t < n) p[t] = v;
}
__global__ void fill_u32(unsigned* p, unsigned v, long long n) {
  long long t = (long long)blockIdx.x * 256 + threadIdx.x;
  if (t < n) p[t] = v;
}
__global__ void unmap_u32(unsigned* p, long long n) {
  long long t = (long long)blockIdx.x * 256 + threadIdx.x;
  if (t < n) { float f = funkey(p[t]); ((float*)p)[t] = f; }
}

// ---------------- KNN (top-6 smallest d2, ties -> lower index) ----------------
__global__ __launch_bounds__(256)
void knn_kernel(const float* __restrict__ pos, int* __restrict__ nbr, int N) {
  __shared__ float sp[256 * 3];
  int b = blockIdx.y;
  int i = blockIdx.x * 256 + threadIdx.x;
  bool act = i < N;
  float px = 0.f, py = 0.f, pz = 0.f;
  if (act) {
    const float* p = pos + ((size_t)b * N + i) * 3;
    px = p[0]; py = p[1]; pz = p[2];
  }
  float bd[KNN_K]; int bi[KNN_K];
#pragma unroll
  for (int q = 0; q < KNN_K; ++q) { bd[q] = 3.4e38f; bi[q] = 0; }
  for (int j0 = 0; j0 < N; j0 += 256) {
    int j = j0 + threadIdx.x;
    if (j < N) {
      const float* p = pos + ((size_t)b * N + j) * 3;
      sp[threadIdx.x * 3 + 0] = p[0];
      sp[threadIdx.x * 3 + 1] = p[1];
      sp[threadIdx.x * 3 + 2] = p[2];
    }
    __syncthreads();
    int cnt = min(256, N - j0);
    if (act) {
      for (int jj = 0; jj < cnt; ++jj) {
        float dx = px - sp[jj * 3 + 0];
        float dy = py - sp[jj * 3 + 1];
        float dz = pz - sp[jj * 3 + 2];
        float d2 = dx * dx + dy * dy + dz * dz;
        if (d2 < bd[KNN_K - 1]) {
          int pp = 0;
#pragma unroll
          for (int q = 0; q < KNN_K - 1; ++q) pp += (bd[q] <= d2) ? 1 : 0;
#pragma unroll
          for (int q = KNN_K - 1; q >= 1; --q)
            if (q > pp) { bd[q] = bd[q - 1]; bi[q] = bi[q - 1]; }
#pragma unroll
          for (int q = 0; q < KNN_K; ++q)
            if (q == pp) { bd[q] = d2; bi[q] = j0 + jj; }
        }
      }
    }
    __syncthreads();
  }
  if (act) {
    int* o = nbr + ((size_t)b * N + i) * KNN_K;
#pragma unroll
    for (int q = 0; q < KNN_K; ++q) o[q] = bi[q];
  }
}

// ------------- edge weights + dedup validity (reverse dup iff i in knn(m)) -------------
__global__ void edge_kernel(const float* __restrict__ pos, const int* __restrict__ nbr,
                            float* __restrict__ wE, unsigned char* __restrict__ validE,
                            const float* __restrict__ eW1, const float* __restrict__ eb1,
                            const float* __restrict__ g, const float* __restrict__ be,
                            const float* __restrict__ mm_, const float* __restrict__ vv_,
                            const float* __restrict__ eW2, const float* __restrict__ eb2,
                            int N) {
  int E = 12 * N;
  long long t = (long long)blockIdx.x * 256 + threadIdx.x;
  if (t >= (long long)BB * E) return;
  int b = (int)(t / E), e = (int)(t % E);
  int NK = N * KNN_K;
  int idx = (e < NK) ? e : e - NK;
  int i = idx / KNN_K;
  int m = nbr[(size_t)b * NK + idx];
  int src, dst; bool val = true;
  if (e < NK) { src = m; dst = i; }
  else {
    src = i; dst = m;
    const int* nm = nbr + ((size_t)b * N + m) * KNN_K;
#pragma unroll
    for (int k = 0; k < KNN_K; ++k) if (nm[k] == i) val = false;
  }
  float w;
  if (src == dst) w = 1.0f;
  else {
    const float* ps = pos + ((size_t)b * N + src) * 3;
    const float* pd = pos + ((size_t)b * N + dst) * 3;
    float dx = ps[0] - pd[0], dy = ps[1] - pd[1], dz = ps[2] - pd[2];
    float d = sqrtf(dx * dx + dy * dy + dz * dz);
    float acc = eb2[0];
#pragma unroll
    for (int c = 0; c < 32; ++c) {
      float u = d * eW1[c] + eb1[c];
      u = (u - mm_[c]) * rsqrtf(vv_[c] + BNEPS) * g[c] + be[c];
      u = fmaxf(u, 0.0f);
      acc += u * eW2[c];
    }
    w = fmaxf(acc, 0.0f);
  }
  if (!val) w = 0.0f;
  wE[(size_t)b * E + e] = w;
  validE[(size_t)b * E + e] = val ? 1 : 0;
}

// ---------------- degree scatter ----------------
__global__ void deg_kernel(const int* __restrict__ nbr, const float* __restrict__ wE,
                           float* __restrict__ deg, int N) {
  int E = 12 * N;
  long long t = (long long)blockIdx.x * 256 + threadIdx.x;
  if (t >= (long long)BB * E) return;
  int b = (int)(t / E), e = (int)(t % E);
  float w = wE[(size_t)b * E + e];
  if (w == 0.0f) return;
  int s, d; decodeEdge(nbr, b, e, N, s, d);
  atomicAdd(&deg[(size_t)b * N + d], w);
}

// ------------- feature scatter: out[dst] += coef * x[src] (coef = w or sym-norm) -------------
__global__ void feat_scatter(const int* __restrict__ nbr, const float* __restrict__ wE,
                             const float* __restrict__ deg, const float* __restrict__ x,
                             float* __restrict__ out, int N) {
  int E = 12 * N;
  long long t = (long long)blockIdx.x * 256 + threadIdx.x;
  long long tot = (long long)BB * E * 64;
  if (t >= tot) return;
  int lane = (int)(t & 63);
  long long eg = t >> 6;
  int b = (int)(eg / E), e = (int)(eg % E);
  float w = wE[(size_t)b * E + e];
  if (w == 0.0f) return;
  int s, d; decodeEdge(nbr, b, e, N, s, d);
  float coef = w;
  if (deg) {
    coef *= rsqrtf(fmaxf(deg[(size_t)b * N + s], 1e-12f)) *
            rsqrtf(fmaxf(deg[(size_t)b * N + d], 1e-12f));
  }
  const float4 xv = *(const float4*)(x + ((size_t)b * N + s) * HB + lane * 4);
  float* o = out + ((size_t)b * N + d) * HB + lane * 4;
  atomicAdd(o + 0, coef * xv.x);
  atomicAdd(o + 1, coef * xv.y);
  atomicAdd(o + 2, coef * xv.z);
  atomicAdd(o + 3, coef * xv.w);
}

// ------------- scatter-max of x_pool[src] into uint-key buffer -------------
__global__ void qmax_scatter(const int* __restrict__ nbr, const unsigned char* __restrict__ validE,
                             const float* __restrict__ xpool, unsigned* __restrict__ qk, int N) {
  int E = 12 * N;
  long long t = (long long)blockIdx.x * 256 + threadIdx.x;
  long long tot = (long long)BB * E * 64;
  if (t >= tot) return;
  int lane = (int)(t & 63);
  long long eg = t >> 6;
  int b = (int)(eg / E), e = (int)(eg % E);
  if (!validE[(size_t)b * E + e]) return;
  int s, d; decodeEdge(nbr, b, e, N, s, d);
  const float4 xv = *(const float4*)(xpool + ((size_t)b * N + s) * HB + lane * 4);
  unsigned* o = qk + ((size_t)b * N + d) * HB + lane * 4;
  atomicMax(o + 0, fkey(xv.x));
  atomicMax(o + 1, fkey(xv.y));
  atomicMax(o + 2, fkey(xv.z));
  atomicMax(o + 3, fkey(xv.w));
}

// ------------- attention score per edge + running max -------------
__global__ void score_kernel(const int* __restrict__ nbr, const unsigned char* __restrict__ validE,
                             const float* __restrict__ xq, const float* __restrict__ xpool,
                             const float* __restrict__ Watt, const float* __restrict__ batt,
                             float* __restrict__ scoreE, unsigned* __restrict__ smaxU, int N) {
  int E = 12 * N;
  long long t = (long long)blockIdx.x * 256 + threadIdx.x;
  if (t >= (long long)BB * E) return;
  int b = (int)(t / E), e = (int)(t % E);
  if (!validE[(size_t)b * E + e]) { scoreE[(size_t)b * E + e] = 0.0f; return; }
  int s, d; decodeEdge(nbr, b, e, N, s, d);
  const float* q = xq + ((size_t)b * N + d) * HB;
  const float* p = xpool + ((size_t)b * N + s) * HB;
  float acc = batt[0];
  for (int c = 0; c < HB; ++c) acc += q[c] * Watt[c];
  for (int c = 0; c < HB; ++c) acc += p[c] * Watt[HB + c];
  acc = (acc > 0.0f) ? acc : NEGS * acc;
  scoreE[(size_t)b * E + e] = acc;
  atomicMax(&smaxU[(size_t)b * N + d], fkey(acc));
}

__global__ void exp_kernel(const int* __restrict__ nbr, const unsigned char* __restrict__ validE,
                           float* __restrict__ scoreE, const unsigned* __restrict__ smaxU,
                           float* __restrict__ esum, int N) {
  int E = 12 * N;
  long long t = (long long)blockIdx.x * 256 + threadIdx.x;
  if (t >= (long long)BB * E) return;
  int b = (int)(t / E), e = (int)(t % E);
  if (!validE[(size_t)b * E + e]) { scoreE[(size_t)b * E + e] = 0.0f; return; }
  int s, d; decodeEdge(nbr, b, e, N, s, d);
  float ev = expf(scoreE[(size_t)b * E + e] - funkey(smaxU[(size_t)b * N + d]));
  scoreE[(size_t)b * E + e] = ev;
  atomicAdd(&esum[(size_t)b * N + d], ev);
}

__global__ void att_scatter(const int* __restrict__ nbr, const unsigned char* __restrict__ validE,
                            const float* __restrict__ scoreE, const float* __restrict__ esum,
                            const float* __restrict__ x, float* __restrict__ xn, int N) {
  int E = 12 * N;
  long long t = (long long)blockIdx.x * 256 + threadIdx.x;
  long long tot = (long long)BB * E * 64;
  if (t >= tot) return;
  int lane = (int)(t & 63);
  long long eg = t >> 6;
  int b = (int)(eg / E), e = (int)(eg % E);
  if (!validE[(size_t)b * E + e]) return;
  int s, d; decodeEdge(nbr, b, e, N, s, d);
  float att = scoreE[(size_t)b * E + e] / esum[(size_t)b * N + d];
  const float4 xv = *(const float4*)(x + ((size_t)b * N + s) * HB + lane * 4);
  float* o = xn + ((size_t)b * N + d) * HB + lane * 4;
  atomicAdd(o + 0, att * xv.x);
  atomicAdd(o + 1, att * xv.y);
  atomicAdd(o + 2, att * xv.z);
  atomicAdd(o + 3, att * xv.w);
}

// ------------- LEConv per-node linear terms -------------
__global__ void nodelin_kernel(const float* __restrict__ xn, const float* __restrict__ lW1,
                               const float* __restrict__ lb1, const float* __restrict__ lW2,
                               const float* __restrict__ lW3, const float* __restrict__ lb3,
                               float* __restrict__ na, float* __restrict__ nbb,
                               float* __restrict__ nl3, int N) {
  long long t = (long long)blockIdx.x * 256 + threadIdx.x;
  if (t >= (long long)BB * N) return;
  const float* x = xn + (size_t)t * HB;
  float a = 0.f, b2 = 0.f, c3 = 0.f;
  for (int c = 0; c < HB; ++c) {
    float v = x[c];
    a += v * lW1[c]; b2 += v * lW2[c]; c3 += v * lW3[c];
  }
  na[t] = a + lb1[0];
  nbb[t] = b2;
  nl3[t] = c3 + lb3[0];
}

__global__ void fitedge_kernel(const int* __restrict__ nbr, const unsigned char* __restrict__ validE,
                               const float* __restrict__ na, const float* __restrict__ nbb,
                               float* __restrict__ fitacc, int N) {
  int E = 12 * N;
  long long t = (long long)blockIdx.x * 256 + threadIdx.x;
  if (t >= (long long)BB * E) return;
  int b = (int)(t / E), e = (int)(t % E);
  if (!validE[(size_t)b * E + e]) return;
  int s, d; decodeEdge(nbr, b, e, N, s, d);
  atomicAdd(&fitacc[(size_t)b * N + d], na[(size_t)b * N + s] - nbb[(size_t)b * N + d]);
}

__global__ void fit_kernel(const float* __restrict__ fitacc, const float* __restrict__ nl3,
                           float* __restrict__ fit, long long n) {
  long long t = (long long)blockIdx.x * 256 + threadIdx.x;
  if (t >= n) return;
  float z = fitacc[t] + nl3[t];
  fit[t] = 1.0f / (1.0f + expf(-z));
}

// ------------- top-k via in-LDS bitonic sort (desc value, asc index on ties) -------------
__global__ __launch_bounds__(512)
void topk_kernel(const float* __restrict__ fit, int* __restrict__ idxO, float* __restrict__ valsO,
                 int N, int P, int kpool) {
  __shared__ unsigned long long sk[2048];
  int b = blockIdx.x;
  for (int t = threadIdx.x; t < P; t += 512) {
    unsigned long long key = 0ull;
    if (t < N) {
      unsigned vk = fkey(fit[(size_t)b * N + t]);
      key = ((unsigned long long)vk << 32) | (unsigned long long)(0xFFFFFFFFu - (unsigned)t);
    }
    sk[t] = key;
  }
  __syncthreads();
  for (int ksz = 2; ksz <= P; ksz <<= 1) {
    for (int j = ksz >> 1; j > 0; j >>= 1) {
      for (int t = threadIdx.x; t < P; t += 512) {
        int ixj = t ^ j;
        if (ixj > t) {
          bool desc = ((t & ksz) == 0);
          unsigned long long a = sk[t], c = sk[ixj];
          bool sw = desc ? (a < c) : (a > c);
          if (sw) { sk[t] = c; sk[ixj] = a; }
        }
      }
      __syncthreads();
    }
  }
  for (int t = threadIdx.x; t < kpool; t += 512) {
    unsigned long long key = sk[t];
    idxO[(size_t)b * 2048 + t] = (int)(0xFFFFFFFFu - (unsigned)(key & 0xFFFFFFFFu));
    valsO[(size_t)b * 2048 + t] = funkey((unsigned)(key >> 32));
  }
}

// ------------- gathers -------------
__global__ void gather_feat(const float* __restrict__ in, float* __restrict__ out,
                            const int* __restrict__ idx, const float* __restrict__ vals,
                            int Nin, int Nout) {
  long long t = (long long)blockIdx.x * 256 + threadIdx.x;
  long long tot = (long long)BB * Nout * HB;
  if (t >= tot) return;
  int c = (int)(t % HB);
  long long r_ = t / HB;
  int r = (int)(r_ % Nout);
  int b = (int)(r_ / Nout);
  int ii = idx[(size_t)b * 2048 + r];
  float v = in[((size_t)b * Nin + ii) * HB + c];
  if (vals) v *= vals[(size_t)b * 2048 + r];
  out[((size_t)b * Nout + r) * HB + c] = v;
}
__global__ void gather_pos(const float* __restrict__ in, float* __restrict__ out,
                           const int* __restrict__ idx, int Nin, int Nout) {
  long long t = (long long)blockIdx.x * 256 + threadIdx.x;
  long long tot = (long long)BB * Nout * 3;
  if (t >= tot) return;
  int c = (int)(t % 3);
  long long r_ = t / 3;
  int r = (int)(r_ % Nout);
  int b = (int)(r_ / Nout);
  int ii = idx[(size_t)b * 2048 + r];
  out[((size_t)b * Nout + r) * 3 + c] = in[((size_t)b * Nin + ii) * 3 + c];
}

// ------------- readout: [mean | max] over nodes -------------
__global__ void readout_kernel(const float* __restrict__ h, float* __restrict__ out,
                               int N, int outBase) {
  int t = blockIdx.x * 256 + threadIdx.x;
  if (t >= BB * HB) return;
  int b = t / HB, c = t % HB;
  float s = 0.0f, mx = -3.4e38f;
  for (int n = 0; n < N; ++n) {
    float v = h[((size_t)b * N + n) * HB + c];
    s += v;
    mx = fmaxf(mx, v);
  }
  out[(size_t)b * 1152 + outBase + c] = s / (float)N;
  out[(size_t)b * 1152 + outBase + 256 + c] = mx;
}

// ------------- final attentional aggregation -------------
__global__ __launch_bounds__(256)
void final_kernel(const float* __restrict__ h, const float* __restrict__ gW,
                  const float* __restrict__ gb, const float* __restrict__ nW,
                  const float* __restrict__ nb_, float* __restrict__ out, int N) {
  __shared__ float gate[64];
  int b = blockIdx.x;
  int t = threadIdx.x;
  if (t < N) {
    float s = gb[0];
    const float* hr = h + ((size_t)b * N + t) * HB;
    for (int c = 0; c < HB; ++c) s += hr[c] * gW[c];
    gate[t] = s;
  }
  __syncthreads();
  if (t == 0) {
    float mx = -3.4e38f;
    for (int n = 0; n < N; ++n) mx = fmaxf(mx, gate[n]);
    float sm = 0.f;
    for (int n = 0; n < N; ++n) { gate[n] = expf(gate[n] - mx); sm += gate[n]; }
    for (int n = 0; n < N; ++n) gate[n] /= sm;
  }
  __syncthreads();
  if (t < 128) {
    float acc = 0.f;
    for (int n = 0; n < N; ++n) {
      const float* hr = h + ((size_t)b * N + n) * HB;
      float d = nb_[t];
      for (int k = 0; k < HB; ++k) d += hr[k] * nW[(size_t)k * 128 + t];
      acc += gate[n] * d;
    }
    out[(size_t)b * 1152 + t] = acc;
  }
}

// ------------- WMMA GEMM: C[M,256] = op(s0*A0 + s1*A1)[M,256] @ W[256,256] (+bias)(+C)(relu) -------------
// Block tile 128x64, 4 waves, each wave 32x64 (two 16x16 row-tiles x four col-tiles).
// LDS tiles are stored in per-lane fragment order so each fragment is one 32B v16h load
// (2x ds_load_b128) instead of 16 scalar ds_load_u16.
//   A rows stored in natural k order; lane<16 reads halves [0..15],
//   lane>=16 reads [16..31] -> contiguous k = 16h + s, same order as B.
//   B stored column-contiguous Bsh[col][k]; lanes 0-15 read k 0-15, lanes 16-31 k 16-31.
__global__ __launch_bounds__(128)
void gemm_wmma(const float* __restrict__ A0, const float* __restrict__ A1,
               float s0, float s1, const float* __restrict__ W,
               const float* __restrict__ bias, float* __restrict__ Cout,
               int M, int flags) {
  __shared__ half_t Ash[128][32];  // natural k order per row
  __shared__ half_t Bsh[64][32];   // [col][k]
  const int tid = threadIdx.x;
  const int wid = tid >> 5;
  const int lane = tid & 31;
  const int rowBase = blockIdx.x * 128;
  const int colBase = blockIdx.y * 64;

  v8f acc[2][4];
#pragma unroll
  for (int h = 0; h < 2; ++h)
#pragma unroll
    for (int j = 0; j < 4; ++j)
#pragma unroll
      for (int e = 0; e < 8; ++e) acc[h][j][e] = 0.0f;

#pragma unroll
  for (int k0 = 0; k0 < 256; k0 += 32) {
    // A tile: 128 rows x 32 k, float4 global loads, natural-order f16 stores
#pragma unroll
    for (int t = tid; t < 128 * 8; t += 128) {
      int r = t >> 3;
      int kq = (t & 7) * 4;                       // 4 consecutive k, same group of 8
      int g = kq >> 3;
      int kd = kq;
      int row = rowBase + r;
      float4 v = make_float4(0.f, 0.f, 0.f, 0.f);
      if (row < M) {
        const float4 a0 = *(const float4*)(A0 + (size_t)row * 256 + k0 + kq);
        v = a0;
        if (A1) {
          const float4 a1 = *(const float4*)(A1 + (size_t)row * 256 + k0 + kq);
          v = make_float4(s0 * a0.x + s1 * a1.x, s0 * a0.y + s1 * a1.y,
                          s0 * a0.z + s1 * a1.z, s0 * a0.w + s1 * a1.w);
        } else {
          v = make_float4(s0 * a0.x, s0 * a0.y, s0 * a0.z, s0 * a0.w);
        }
      }
      Ash[r][kd + 0] = (half_t)v.x;
      Ash[r][kd + 1] = (half_t)v.y;
      Ash[r][kd + 2] = (half_t)v.z;
      Ash[r][kd + 3] = (half_t)v.w;
    }
    // B tile: 32 k x 64 cols, coalesced global reads, column-contiguous stores
#pragma unroll
    for (int t = tid; t < 32 * 64; t += 128) {
      int kk = t >> 6, c = t & 63;
      Bsh[c][kk] = (half_t)W[(size_t)(k0 + kk) * 256 + colBase + c];
    }
    __syncthreads();

    const int hoff = (lane < 16) ? 0 : 16;      // fragment half-select (A and B natural)
    const v16h a0f = *(const v16h*)&Ash[wid * 32 + (lane & 15)][hoff];
    const v16h a1f = *(const v16h*)&Ash[wid * 32 + 16 + (lane & 15)][hoff];
#pragma unroll
    for (int j = 0; j < 4; ++j) {
      const v16h bf = *(const v16h*)&Bsh[j * 16 + (lane & 15)][hoff];
      acc[0][j] = __builtin_amdgcn_wmma_f32_16x16x32_f16(false, a0f, false, bf, (short)0,
                                                         acc[0][j], false, false);
      acc[1][j] = __builtin_amdgcn_wmma_f32_16x16x32_f16(false, a1f, false, bf, (short)0,
                                                         acc[1][j], false, false);
    }
    __syncthreads();
  }

  const int rofs = (lane < 16) ? 0 : 8;
#pragma unroll
  for (int h = 0; h < 2; ++h) {
#pragma unroll
    for (int j = 0; j < 4; ++j) {
      const int coln = colBase + j * 16 + (lane & 15);
#pragma unroll
      for (int e = 0; e < 8; ++e) {
        int row = rowBase + wid * 32 + h * 16 + rofs + e;
        if (row < M) {
          float v = acc[h][j][e];
          if (bias) v += bias[coln];
          size_t o = (size_t)row * 256 + coln;
          if (flags & 2) v += Cout[o];
          if (flags & 1) v = fmaxf(v, 0.0f);
          Cout[o] = v;
        }
      }
    }
  }
}

// =================== host orchestration ===================
static inline unsigned blks(long long n) { return (unsigned)((n + 255) / 256); }

extern "C" void kernel_launch(void* const* d_in, const int* in_sizes, int n_in,
                              void* d_out, int out_size, void* d_ws, size_t ws_size,
                              hipStream_t stream) {
  (void)in_sizes; (void)n_in; (void)out_size; (void)ws_size;
  const float* x_in   = (const float*)d_in[0];
  const float* pos_in = (const float*)d_in[1];
  const float* conv_W = (const float*)d_in[2];
  const float* eW1 = (const float*)d_in[3];
  const float* eb1 = (const float*)d_in[4];
  const float* ebn_g = (const float*)d_in[5];
  const float* ebn_b = (const float*)d_in[6];
  const float* ebn_m = (const float*)d_in[7];
  const float* ebn_v = (const float*)d_in[8];
  const float* eW2 = (const float*)d_in[9];
  const float* eb2 = (const float*)d_in[10];
  const float* aWrel = (const float*)d_in[11];
  const float* abrel = (const float*)d_in[12];
  const float* aWroot = (const float*)d_in[13];
  const float* aWlin = (const float*)d_in[14];
  const float* ablin = (const float*)d_in[15];
  const float* aWatt = (const float*)d_in[16];
  const float* abatt = (const float*)d_in[17];
  const float* leW1 = (const float*)d_in[18];
  const float* leb1 = (const float*)d_in[19];
  const float* leW2 = (const float*)d_in[20];
  const float* leW3 = (const float*)d_in[21];
  const float* leb3 = (const float*)d_in[22];
  const float* gW = (const float*)d_in[23];
  const float* gb = (const float*)d_in[24];
  const float* nW = (const float*)d_in[25];
  const float* nbv = (const float*)d_in[26];
  float* out = (float*)d_out;

  char* ws = (char*)d_ws;
  size_t off = 0;
  auto take = [&](size_t bytes) -> char* {
    char* p = ws + off;
    off += (bytes + 255) & ~(size_t)255;
    return p;
  };
  const size_t BIGN = (size_t)BB * 2048 * HB;
  float* big[5];
  for (int i = 0; i < 5; ++i) big[i] = (float*)take(BIGN * 4);
  float* posA = (float*)take((size_t)BB * 2048 * 3 * 4);
  float* posB = (float*)take((size_t)BB * 2048 * 3 * 4);
  int* nbr = (int*)take((size_t)BB * 2048 * 6 * 4);
  float* wE = (float*)take((size_t)BB * 24576 * 4);
  unsigned char* validE = (unsigned char*)take((size_t)BB * 24576);
  float* scoreE = (float*)take((size_t)BB * 24576 * 4);
  float* deg = (float*)take((size_t)BB * 2048 * 4);
  unsigned* smaxU = (unsigned*)take((size_t)BB * 2048 * 4);
  float* esum = (float*)take((size_t)BB * 2048 * 4);
  float* fitacc = (float*)take((size_t)BB * 2048 * 4);
  float* fitv = (float*)take((size_t)BB * 2048 * 4);
  float* na = (float*)take((size_t)BB * 2048 * 4);
  float* nbb = (float*)take((size_t)BB * 2048 * 4);
  float* nl3 = (float*)take((size_t)BB * 2048 * 4);
  int* idxP = (int*)take((size_t)BB * 2048 * 4);
  float* valsP = (float*)take((size_t)BB * 2048 * 4);

  float* pH = big[0];
  float* pT0 = big[1];
  float* pT1 = big[2];
  float* pT2 = big[3];
  float* xc_buf = nullptr;          // stage 0 reads x directly from input
  const float* xc_read = x_in;
  const float* pos_read = pos_in;

  const int Ns[4] = {2048, 308, 77, 39};
  const int Ps[3] = {2048, 512, 128};
  const int Kp[3] = {308, 77, 39};

  for (int s = 0; s < 4; ++s) {
    int N = Ns[s];
    int E = 12 * N;
    int M = BB * N;
    long long MH = (long long)M * HB;
    long long totE = (long long)BB * E;
    long long tot64 = totE * 64;
    dim3 ggrid((M + 127) / 128, 4);

    knn_kernel<<<dim3((N + 255) / 256, BB), 256, 0, stream>>>(pos_read, nbr, N);
    edge_kernel<<<blks(totE), 256, 0, stream>>>(pos_read, nbr, wE, validE,
        eW1 + s * 32, eb1 + s * 32, ebn_g + s * 32, ebn_b + s * 32,
        ebn_m + s * 32, ebn_v + s * 32, eW2 + s * 32, eb2 + s, N);

    if (s == 1) readout_kernel<<<BB, 256, 0, stream>>>(pH, out, N, 640);
    if (s == 2) readout_kernel<<<BB, 256, 0, stream>>>(pH, out, N, 128);

    // GCN2Conv: agg
    fill_f32<<<blks((long long)BB * N), 256, 0, stream>>>(deg, 0.f, (long long)BB * N);
    deg_kernel<<<blks(totE), 256, 0, stream>>>(nbr, wE, deg, N);
    fill_f32<<<blks(MH), 256, 0, stream>>>(pT0, 0.f, MH);
    const float* hin = (s == 0) ? x_in : pH;
    feat_scatter<<<blks(tot64), 256, 0, stream>>>(nbr, wE, deg, hin, pT0, N);
    // h = relu((0.8*agg + 0.2*x0) @ conv_W[s])
    gemm_wmma<<<ggrid, 128, 0, stream>>>(pT0, xc_read, 0.8f, 0.2f,
        conv_W + (size_t)s * 65536, nullptr, pH, M, 1);

    if (s == 3) break;

    // ----- ASAP pooling -----
    fill_f32<<<blks(MH), 256, 0, stream>>>(pT0, 0.f, MH);
    feat_scatter<<<blks(tot64), 256, 0, stream>>>(nbr, wE, nullptr, pH, pT0, N);
    // x_pool = sc @ Wrel + brel + h @ Wroot
    gemm_wmma<<<ggrid, 128, 0, stream>>>(pT0, nullptr, 1.f, 0.f,
        aWrel + (size_t)s * 65536, abrel + s * 256, pT1, M, 0);
    gemm_wmma<<<ggrid, 128, 0, stream>>>(pH, nullptr, 1.f, 0.f,
        aWroot + (size_t)s * 65536, nullptr, pT1, M, 2);
    // x_q = scatter_max(x_pool[src]) @ Wlin + blin
    fill_u32<<<blks(MH), 256, 0, stream>>>((unsigned*)pT2, NEGINF_KEY, MH);
    qmax_scatter<<<blks(tot64), 256, 0, stream>>>(nbr, validE, pT1, (unsigned*)pT2, N);
    unmap_u32<<<blks(MH), 256, 0, stream>>>((unsigned*)pT2, MH);
    gemm_wmma<<<ggrid, 128, 0, stream>>>(pT2, nullptr, 1.f, 0.f,
        aWlin + (size_t)s * 65536, ablin + s * 256, pT0, M, 0);
    // per-dst softmax attention
    fill_u32<<<blks((long long)BB * N), 256, 0, stream>>>(smaxU, NEGINF_KEY, (long long)BB * N);
    score_kernel<<<blks(totE), 256, 0, stream>>>(nbr, validE, pT0, pT1,
        aWatt + (size_t)s * 512, abatt + s, scoreE, smaxU, N);
    fill_f32<<<blks((long long)BB * N), 256, 0, stream>>>(esum, 0.f, (long long)BB * N);
    exp_kernel<<<blks(totE), 256, 0, stream>>>(nbr, validE, scoreE, smaxU, esum, N);
    fill_f32<<<blks(MH), 256, 0, stream>>>(pT2, 0.f, MH);
    att_scatter<<<blks(tot64), 256, 0, stream>>>(nbr, validE, scoreE, esum, pH, pT2, N);
    // LEConv fitness
    nodelin_kernel<<<blks((long long)BB * N), 256, 0, stream>>>(pT2, leW1 + s * 256, leb1 + s,
        leW2 + s * 256, leW3 + s * 256, leb3 + s, na, nbb, nl3, N);
    fill_f32<<<blks((long long)BB * N), 256, 0, stream>>>(fitacc, 0.f, (long long)BB * N);
    fitedge_kernel<<<blks(totE), 256, 0, stream>>>(nbr, validE, na, nbb, fitacc, N);
    fit_kernel<<<blks((long long)BB * N), 256, 0, stream>>>(fitacc, nl3, fitv, (long long)BB * N);
    // top-k + gathers
    topk_kernel<<<BB, 512, 0, stream>>>(fitv, idxP, valsP, N, Ps[s], Kp[s]);
    int Nn = Kp[s];
    long long totg = (long long)BB * Nn * HB;
    gather_feat<<<blks(totg), 256, 0, stream>>>(pT2, pH, idxP, valsP, N, Nn);
    float* xdst = (xc_buf == nullptr) ? big[4] : pT1;
    gather_feat<<<blks(totg), 256, 0, stream>>>(xc_read, xdst, idxP, nullptr, N, Nn);
    if (xc_buf == nullptr) { xc_buf = xdst; }
    else { float* tmp = xc_buf; xc_buf = xdst; pT1 = tmp; }
    xc_read = xc_buf;
    float* pdst = (pos_read == pos_in) ? posA : ((pos_read == posA) ? posB : posA);
    gather_pos<<<blks((long long)BB * Nn * 3), 256, 0, stream>>>(pos_read, pdst, idxP, N, Nn);
    pos_read = pdst;
  }

  final_kernel<<<BB, 256, 0, stream>>>(pH, gW, gb, nW, nbv, out, 39);
}
